// SwinAttention_60593398612622
// MI455X (gfx1250) — matrix operand order repaired and checked by reading, not compiled, not run on hardware
//
#include <hip/hip_runtime.h>
#include <hip/hip_bf16.h>

typedef __attribute__((ext_vector_type(16))) _Float16 v16h;
typedef __attribute__((ext_vector_type(8)))  _Float16 v8h;
typedef __attribute__((ext_vector_type(8)))  float    v8f;
typedef __attribute__((ext_vector_type(4)))  float    v4f;

#define DIM_    192
#define HEADS_  6
#define HD_     32
#define WS_     8
#define SS_     4
#define NTOK    64           // tokens per window
#define BATCH   4
#define IMH     256
#define IMW     256
#define NWX     32           // windows per row
#define NWIN    1024         // windows per image
#define SCALE_  0.17677669529663687f   // HD^-0.5

#define KSTEPS      6        // 192 / 32
#define QKV_TILES_N 36       // 576 / 16
#define PRJ_TILES_N 12       // 192 / 16
#define WSQ_HALFS   (QKV_TILES_N * KSTEPS * 32 * 16)   // 110592 f16
#define WSP_HALFS   (PRJ_TILES_N * KSTEPS * 32 * 16)   //  36864 f16

// ---- LDS arena layout (bytes); lO (f32, 48KB) aliases lx+lQ (dead then) ----
#define OFF_LX   0            // 24576  window input f16, token-major
#define OFF_LQ   24576        // 24576  Q*scale f16, row-major per head
#define OFF_LKF  49152        // 24576  K in WMMA-B fragment order
#define OFF_LVF  73728        // 24576  V in WMMA-B fragment order
#define OFF_LS   98304        // 16384  logits f32 (one head)
#define OFF_LP   114688       //  8192  probs f16 (one head)
#define OFF_LY   122880       // 24576  attn output f16, token-major
#define OFF_RPB  147456       //  5400  rel-pos bias table f32
#define OFF_RED  152856       //  1024  softmax reduction buffer
#define SMEM_SZ  153880       // ~150 KB -> 2 workgroups per 320KB WGP

__device__ __forceinline__ v16h make_frag(v8h lo, v8h hi) {
  v16h r;
#pragma unroll
  for (int i = 0; i < 8; ++i) { r[i] = lo[i]; r[i + 8] = hi[i]; }
  return r;
}

__device__ __forceinline__ v8f wmma_f16(v16h a, v16h b, v8f c) {
  return __builtin_amdgcn_wmma_f32_16x16x32_f16(false, a, false, b,
                                                (short)0, c, false, false);
}

// ---------------------------------------------------------------------------
// Kernel 1: swizzle qkv_w (576x192) and proj_w (192x192) into f16 WMMA-B
// fragment order:  ws[((tn*6+ks)*32 + lane)*16 + i] = W[n][k]
//   with n = tn*16 + (lane&15), k = ks*32 + (lane>>4)*16 + i
// so each lane's B fragment is one contiguous 32-byte chunk.
// ---------------------------------------------------------------------------
__global__ void swin_wconv(const float* __restrict__ qkv_w,
                           const float* __restrict__ proj_w,
                           _Float16* __restrict__ wsq,
                           _Float16* __restrict__ wsp) {
  int idx = blockIdx.x * 256 + threadIdx.x;
  if (idx < WSQ_HALFS) {
    int i    = idx & 15;
    int lane = (idx >> 4) & 31;
    int rest = idx >> 9;
    int ks   = rest % KSTEPS;
    int tn   = rest / KSTEPS;
    int k = ks * 32 + (lane >> 4) * 16 + i;      // 0..191
    int n = tn * 16 + (lane & 15);               // 0..575
    wsq[idx] = (_Float16)qkv_w[n * DIM_ + k];
  }
  if (idx < WSP_HALFS) {
    int i    = idx & 15;
    int lane = (idx >> 4) & 31;
    int rest = idx >> 9;
    int ks   = rest % KSTEPS;
    int tn   = rest / KSTEPS;
    int k = ks * 32 + (lane >> 4) * 16 + i;      // 0..191
    int n = tn * 16 + (lane & 15);               // 0..191
    wsp[idx] = (_Float16)proj_w[n * DIM_ + k];
  }
}

// ---------------------------------------------------------------------------
// Kernel 2: fully fused window attention, one block (8 wave32) per window.
// ---------------------------------------------------------------------------
__global__ void __launch_bounds__(256)
swin_attn(const float* __restrict__ x,        // (B,192,256,256)
          const float* __restrict__ qkv_b,    // (576)
          const float* __restrict__ proj_b,   // (192)
          const float* __restrict__ rpb,      // (225,6)
          const _Float16* __restrict__ wsq,   // swizzled qkv weights
          const _Float16* __restrict__ wsp,   // swizzled proj weights
          float* __restrict__ out) {          // (B,192,256,256)
  __shared__ __align__(16) unsigned char smem[SMEM_SZ];
  _Float16* lx   = (_Float16*)(smem + OFF_LX);
  _Float16* lQ   = (_Float16*)(smem + OFF_LQ);   // [head][64*32]
  _Float16* lKf  = (_Float16*)(smem + OFF_LKF);  // [head][4*32*16]
  _Float16* lVf  = (_Float16*)(smem + OFF_LVF);  // [head][4*32*16]
  float*    lS   = (float*)   (smem + OFF_LS);
  _Float16* lP   = (_Float16*)(smem + OFF_LP);
  _Float16* lY   = (_Float16*)(smem + OFF_LY);
  float*    lrpb = (float*)   (smem + OFF_RPB);
  float*    lred = (float*)   (smem + OFF_RED);
  float*    lO   = (float*)   (smem + OFF_LX);   // aliases lx+lQ (phase D)

  const int tid   = threadIdx.x;
  const int lane  = tid & 31;
  const int wave  = tid >> 5;
  const int m     = lane & 15;     // row/col within 16x16 tile fragments
  const int khalf = lane >> 4;     // lane half (fragment K offset select)

  const int bwin = blockIdx.x;
  const int b  = bwin / NWIN;
  const int nw = bwin % NWIN;
  const int wy = nw / NWX, wx = nw % NWX;
  const bool swy = (wy == NWX - 1), swx = (wx == NWX - 1); // boundary windows

  // ---- Phase A: stage rel-pos table + rolled/shifted window into LDS ------
  // Every 4-pixel quad is contiguous even across the roll wrap: the base
  // (wx*8 + q*4 + SS) & 255 is 4-aligned, so quads never straddle 256.
  for (int i = tid; i < 225 * HEADS_; i += 256) lrpb[i] = rpb[i];
  for (int i = tid; i < DIM_ * 8 * 2; i += 256) {     // 3072 float4 quads
    int q  = i & 1;                 // which quad of the 8-px row
    int ty = (i >> 1) & 7;          // window row
    int c  = i >> 4;                // channel 0..191
    int gy  = (wy * WS_ + ty + SS_) & (IMH - 1);
    int gxb = (wx * WS_ + q * 4 + SS_) & (IMW - 1);
    v4f v = *(const v4f*)&x[((b * DIM_ + c) * IMH + gy) * IMW + gxb];
#pragma unroll
    for (int j = 0; j < 4; ++j) {
      int t = ty * 8 + q * 4 + j;
      lx[t * DIM_ + c] = (_Float16)v[j];
    }
  }
  __syncthreads();

  // ---- Phase B: QKV GEMM (64x192)@(192x576) -------------------------------
  // Wave's token tile tm = wave&3 is constant -> hoist all 6 A fragments.
  // Split into three branch-free loops (Q cols, K cols, V cols).
  {
    const int tmB = wave & 3;
    const int tn0 = wave >> 2;     // 0 or 1; step 2 -> 6 tiles per group
    v16h af[KSTEPS];
#pragma unroll
    for (int ks = 0; ks < KSTEPS; ++ks) {
      const _Float16* ap = &lx[(tmB * 16 + m) * DIM_ + ks * 32 + khalf * 16];
      af[ks] = make_frag(*(const v8h*)ap, *(const v8h*)(ap + 8));
    }
    // --- Q columns (tn 0..11) -> lQ row-major, pre-scaled ---
    for (int tn = tn0; tn < 12; tn += 2) {
      v8f acc = {};
#pragma unroll
      for (int ks = 0; ks < KSTEPS; ++ks) {
        const _Float16* bp = wsq + (((tn * KSTEPS + ks) * 32 + lane) << 4);
        acc = wmma_f16(af[ks], make_frag(*(const v8h*)bp, *(const v8h*)(bp + 8)), acc);
      }
      int hh = tn >> 1, dhal = tn & 1;
      float bias = qkv_b[tn * 16 + m];
#pragma unroll
      for (int r = 0; r < 8; ++r) {
        int tok = tmB * 16 + khalf * 8 + r;
        lQ[hh * 2048 + tok * HD_ + dhal * 16 + m] =
            (_Float16)((acc[r] + bias) * SCALE_);
      }
    }
    // --- K columns (tn 12..23) -> lKf in WMMA-B fragment order ---
    for (int tn = tn0; tn < 12; tn += 2) {
      v8f acc = {};
#pragma unroll
      for (int ks = 0; ks < KSTEPS; ++ks) {
        const _Float16* bp = wsq + ((((tn + 12) * KSTEPS + ks) * 32 + lane) << 4);
        acc = wmma_f16(af[ks], make_frag(*(const v8h*)bp, *(const v8h*)(bp + 8)), acc);
      }
      int hh = tn >> 1, dhal = tn & 1;
      float bias = qkv_b[(tn + 12) * 16 + m];
      _Float16* kp = &lKf[hh * 2048 + (tmB * 32 + dhal * 16 + khalf * 8) * 16 + m];
#pragma unroll
      for (int r = 0; r < 8; ++r) kp[r * 16] = (_Float16)(acc[r] + bias);
    }
    // --- V columns (tn 24..35) -> lVf, 8 contiguous halves = 1 b128 store ---
    for (int tn = tn0; tn < 12; tn += 2) {
      v8f acc = {};
#pragma unroll
      for (int ks = 0; ks < KSTEPS; ++ks) {
        const _Float16* bp = wsq + ((((tn + 24) * KSTEPS + ks) * 32 + lane) << 4);
        acc = wmma_f16(af[ks], make_frag(*(const v8h*)bp, *(const v8h*)(bp + 8)), acc);
      }
      int hh = tn >> 1, dhal = tn & 1;
      float bias = qkv_b[(tn + 24) * 16 + m];
      v8h pk;
#pragma unroll
      for (int r = 0; r < 8; ++r) pk[r] = (_Float16)(acc[r] + bias);
      *(v8h*)&lVf[hh * 2048 +
                  ((((tmB >> 1) * 2 + dhal) * 32) + (tmB & 1) * 16 + m) * 16 +
                  khalf * 8] = pk;
    }
  }
  __syncthreads();

  // ---- Phase C: per-head attention ---------------------------------------
  for (int hh = 0; hh < HEADS_; ++hh) {
    // C1: S = Q @ K^T  (64x64, K=HD=32 -> one WMMA per tile) + bias + mask
    for (int tile = wave; tile < 16; tile += 8) {
      int tm = tile >> 2, tn = tile & 3;
      const _Float16* ap = &lQ[hh * 2048 + (tm * 16 + m) * HD_ + khalf * 16];
      v16h a = make_frag(*(const v8h*)ap, *(const v8h*)(ap + 8));
      const _Float16* bp = &lKf[hh * 2048 + ((tn * 32 + lane) << 4)];
      v16h bb = make_frag(*(const v8h*)bp, *(const v8h*)(bp + 8));
      v8f acc = {};
      acc = wmma_f16(a, bb, acc);
      int col = tn * 16 + m;               // key token
      int ky = col >> 3, kx = col & 7;
      int rk  = (swy ? (ky >= SS_) + 1 : 0) * 3 + (swx ? (kx >= SS_) + 1 : 0);
      int krp = ky * 15 + kx;              // key part of rel-pos index
#pragma unroll
      for (int r = 0; r < 8; ++r) {
        int row = tm * 16 + khalf * 8 + r; // query token
        int qy = row >> 3, qx = row & 7;
        int rpi = (qy * 15 + qx) - krp + 112;   // (dy+7)*15 + (dx+7)
        float bv = lrpb[rpi * HEADS_ + hh];
        int rq = (swy ? (qy >= SS_) + 1 : 0) * 3 + (swx ? (qx >= SS_) + 1 : 0);
        float mk = (rq != rk) ? -100.0f : 0.0f;
        lS[row * NTOK + col] = acc[r] + bv + mk;
      }
    }
    __syncthreads();

    // C2: row softmax -> P (f16).  4 threads per row, 16 cols each.
    {
      int row = tid & 63;
      int seg = tid >> 6;                  // 0..3
      const float* srow = &lS[row * NTOK + seg * 16];
      float mx = -1e30f;
#pragma unroll
      for (int j = 0; j < 16; ++j) mx = fmaxf(mx, srow[j]);
      lred[seg * 64 + row] = mx;
      __syncthreads();
      mx = fmaxf(fmaxf(lred[row], lred[64 + row]),
                 fmaxf(lred[128 + row], lred[192 + row]));
      float sum = 0.0f;
#pragma unroll
      for (int j = 0; j < 16; ++j) sum += __expf(srow[j] - mx);
      __syncthreads();
      lred[seg * 64 + row] = sum;
      __syncthreads();
      sum = lred[row] + lred[64 + row] + lred[128 + row] + lred[192 + row];
      float inv = 1.0f / sum;
      _Float16* prow = &lP[row * NTOK + seg * 16];
#pragma unroll
      for (int j = 0; j < 16; ++j)
        prow[j] = (_Float16)(__expf(srow[j] - mx) * inv);
    }
    __syncthreads();

    // C3: Y_h = P @ V  (64x32, K=64 -> 2 WMMA steps; 8 tiles, 1 per wave)
    {
      int tm = wave >> 1, tn = wave & 1;
      v8f acc = {};
#pragma unroll
      for (int ks = 0; ks < 2; ++ks) {
        const _Float16* ap = &lP[(tm * 16 + m) * NTOK + ks * 32 + khalf * 16];
        v16h a = make_frag(*(const v8h*)ap, *(const v8h*)(ap + 8));
        const _Float16* bp = &lVf[hh * 2048 + (((ks * 2 + tn) * 32 + lane) << 4)];
        v16h bb = make_frag(*(const v8h*)bp, *(const v8h*)(bp + 8));
        acc = wmma_f16(a, bb, acc);
      }
      int col = tn * 16 + m;
#pragma unroll
      for (int r = 0; r < 8; ++r) {
        int row = tm * 16 + khalf * 8 + r;
        lY[row * DIM_ + hh * HD_ + col] = (_Float16)acc[r];
      }
    }
    __syncthreads();
  }

  // ---- Phase D: proj GEMM (64x192)@(192x192) -> channel-major f32 in LDS --
  // tm is wave-constant again: hoist the 6 A fragments from lY.
  {
    const int tmD = wave & 3;
    v16h af[KSTEPS];
#pragma unroll
    for (int ks = 0; ks < KSTEPS; ++ks) {
      const _Float16* ap = &lY[(tmD * 16 + m) * DIM_ + ks * 32 + khalf * 16];
      af[ks] = make_frag(*(const v8h*)ap, *(const v8h*)(ap + 8));
    }
    for (int tn = (wave >> 2); tn < 12; tn += 2) {
      v8f acc = {};
#pragma unroll
      for (int ks = 0; ks < KSTEPS; ++ks) {
        const _Float16* bp = wsp + (((tn * KSTEPS + ks) * 32 + lane) << 4);
        acc = wmma_f16(af[ks], make_frag(*(const v8h*)bp, *(const v8h*)(bp + 8)), acc);
      }
      int col = tn * 16 + m;               // channel 0..191
      float bias = proj_b[col];
      // channel-major staging: the 8 accumulator rows are contiguous tokens
      v4f lo, hi;
#pragma unroll
      for (int r = 0; r < 4; ++r) { lo[r] = acc[r] + bias; hi[r] = acc[r + 4] + bias; }
      float* op = &lO[col * NTOK + tmD * 16 + khalf * 8];
      *(v4f*)op = lo;
      *(v4f*)(op + 4) = hi;
    }
  }
  __syncthreads();

  // ---- Phase E: coalesced float4 output with reverse roll -----------------
  for (int i = tid; i < DIM_ * 8 * 2; i += 256) {     // 3072 float4 quads
    int q  = i & 1;
    int ty = (i >> 1) & 7;
    int c  = i >> 4;
    int gy  = (wy * WS_ + ty + SS_) & (IMH - 1);
    int gxb = (wx * WS_ + q * 4 + SS_) & (IMW - 1);
    v4f vv = *(const v4f*)&lO[c * NTOK + ty * 8 + q * 4];
    *(v4f*)&out[((b * DIM_ + c) * IMH + gy) * IMW + gxb] = vv;
  }
}

// ---------------------------------------------------------------------------
extern "C" void kernel_launch(void* const* d_in, const int* in_sizes, int n_in,
                              void* d_out, int out_size, void* d_ws, size_t ws_size,
                              hipStream_t stream) {
  const float* x      = (const float*)d_in[0];
  const float* qkv_w  = (const float*)d_in[1];
  const float* qkv_b  = (const float*)d_in[2];
  const float* proj_w = (const float*)d_in[3];
  const float* proj_b = (const float*)d_in[4];
  const float* rpb    = (const float*)d_in[5];
  float* out = (float*)d_out;

  _Float16* wsq = (_Float16*)d_ws;
  _Float16* wsp = wsq + WSQ_HALFS;

  swin_wconv<<<(WSQ_HALFS + 255) / 256, 256, 0, stream>>>(qkv_w, proj_w, wsq, wsp);
  swin_attn<<<BATCH * NWIN, 256, 0, stream>>>(x, qkv_b, proj_b, rpb, wsq, wsp, out);
}
